// HierarchicalGNNBlock_7559142441639
// MI455X (gfx1250) — compile-verified, matrix operand order checked by reading, not run on hardware
//
#include <hip/hip_runtime.h>

#define NN 50000
#define NE 400000
#define NS 256

typedef float v8f __attribute__((ext_vector_type(8)));
typedef __bf16 v16bf __attribute__((ext_vector_type(16)));

// dynamic LDS sizes (bytes)
#define MLP_SMEM          222208
#define ATTN256_SMEM      136960
#define ATTN_STREAM_SMEM  117504

static __device__ __forceinline__ unsigned short f2bf(float f) {
  unsigned u = __float_as_uint(f);
  unsigned r = u + 0x7fffu + ((u >> 16) & 1u);
  return (unsigned short)(r >> 16);
}

// pack two floats into two bf16 in one dword
static __device__ __forceinline__ unsigned pk2(float a, float b) {
  return (unsigned)f2bf(a) | ((unsigned)f2bf(b) << 16);
}

static __device__ __forceinline__ v8f v8f_zero() {
  v8f z = {0.f, 0.f, 0.f, 0.f, 0.f, 0.f, 0.f, 0.f};
  return z;
}

static __device__ __forceinline__ v8f wmma_bf16(v16bf a, v16bf b, v8f c) {
  // D = A(16x32 bf16) x B(32x16 bf16) + C(16x16 f32)
  return __builtin_amdgcn_wmma_f32_16x16x32_bf16(false, a, false, b, (short)0, c,
                                                 false, false);
}

// Load a 16x32 bf16 operand fragment from LDS, [row][k] storage, strideH halves.
static __device__ __forceinline__ v16bf load_frag(const unsigned short* p, int strideH,
                                                  int lane) {
  int r = lane & 15;
  int k = (lane & 16) ? 8 : 0;
  const uint4* q0 = (const uint4*)(p + r * strideH + k);
  const uint4* q1 = (const uint4*)(p + r * strideH + k + 16);
  union { v16bf v; uint4 u[2]; } f;
  f.u[0] = *q0;
  f.u[1] = *q1;
  return f.v;
}

// ---------------------------------------------------------------------------
// f32 -> bf16 bulk conversion (weights), vectorized float4 -> b64
// ---------------------------------------------------------------------------
__global__ void gnn_cvt_bf16(const float* __restrict__ s, unsigned short* __restrict__ d,
                             int n4) {
  int i = blockIdx.x * 256 + threadIdx.x;
  if (i < n4) {
    float4 v = ((const float4*)s)[i];
    uint2 r;
    r.x = pk2(v.x, v.y);
    r.y = pk2(v.z, v.w);
    ((uint2*)d)[i] = r;
  }
}

// ---------------------------------------------------------------------------
// proj128: C[M,128] = (A[M,128] @ W^T[128,128] + b) * scale   (W pre-bf16)
// 128-row blocks; wave tile 32x64 -> 8 WMMA per barrier pair.
// ---------------------------------------------------------------------------
__global__ void gnn_proj128(const float* __restrict__ A,
                            const unsigned short* __restrict__ Wb,
                            const float* __restrict__ bias, float* __restrict__ C,
                            int M, float scale) {
  __shared__ unsigned short sA[128 * 136];
  __shared__ unsigned short sW[128 * 40];
  const int tid = threadIdx.x, lane = tid & 31, wid = tid >> 5;
  const int m0 = blockIdx.x * 128;
  for (int idx = tid; idx < 4096; idx += 256) {
    int r = idx >> 5, c4 = idx & 31;
    int row = m0 + r;
    float4 v = row < M ? ((const float4*)(A + (size_t)row * 128))[c4]
                       : make_float4(0.f, 0.f, 0.f, 0.f);
    uint2 p; p.x = pk2(v.x, v.y); p.y = pk2(v.z, v.w);
    *(uint2*)(sA + r * 136 + c4 * 4) = p;
  }
  const int wM = wid >> 1, wN = wid & 1;
  v8f acc[2][4] = {{v8f_zero(), v8f_zero(), v8f_zero(), v8f_zero()},
                   {v8f_zero(), v8f_zero(), v8f_zero(), v8f_zero()}};
  for (int k0 = 0; k0 < 128; k0 += 32) {
    __syncthreads();
    for (int idx = tid; idx < 512; idx += 256) {
      int nl = idx >> 2, q = idx & 3;
      ((uint4*)(sW + nl * 40))[q] = ((const uint4*)(Wb + (size_t)nl * 128 + k0))[q];
    }
    __syncthreads();
    v16bf a0 = load_frag(sA + (wM * 32) * 136 + k0, 136, lane);
    v16bf a1 = load_frag(sA + (wM * 32 + 16) * 136 + k0, 136, lane);
#pragma unroll
    for (int nt = 0; nt < 4; ++nt) {
      v16bf b = load_frag(sW + (wN * 64 + nt * 16) * 40, 40, lane);
      acc[0][nt] = wmma_bf16(a0, b, acc[0][nt]);
      acc[1][nt] = wmma_bf16(a1, b, acc[1][nt]);
    }
  }
  const int hi8 = (lane >> 4) * 8;
#pragma unroll
  for (int mt = 0; mt < 2; ++mt)
#pragma unroll
    for (int nt = 0; nt < 4; ++nt)
#pragma unroll
      for (int i = 0; i < 8; ++i) {
        int r = wM * 32 + mt * 16 + i + hi8;
        int c = wN * 64 + nt * 16 + (lane & 15);
        int row = m0 + r;
        if (row < M) C[(size_t)row * 128 + c] = (acc[mt][nt][i] + bias[c]) * scale;
      }
}

// ---------------------------------------------------------------------------
// Fused 2-layer MLP (hidden [64,512] never leaves LDS). W1b/W2b pre-bf16.
// concat row m = [X0[i0[m]], X1[i1[m]], X2[m]]
// ---------------------------------------------------------------------------
__global__ void gnn_mlp_fused(const float* __restrict__ X0, const float* __restrict__ X1,
                              const float* __restrict__ X2, const int* __restrict__ idx0,
                              const int* __restrict__ idx1,
                              const unsigned short* __restrict__ W1b,
                              const float* __restrict__ b1, const float* __restrict__ g1,
                              const float* __restrict__ be1,
                              const unsigned short* __restrict__ W2b,
                              const float* __restrict__ b2, const float* __restrict__ g2,
                              const float* __restrict__ be2,
                              const float* __restrict__ resid, float* __restrict__ Out,
                              int M, int act) {
  extern __shared__ char smem[];
  unsigned short* sX = (unsigned short*)smem;              // [64][392] -> sHb [64][520]
  float* sF = (float*)(smem + 66560);                      // [64][516] -> [64][136]
  unsigned short* sW = (unsigned short*)(smem + 198656);   // [256][40]
  float* red = (float*)(smem + 219136);                    // 512 f
  float* rstat = (float*)(smem + 221184);                  // 128 f
  int* sI = (int*)(smem + 221696);                         // 128 i
  const int tid = threadIdx.x, lane = tid & 31, wid = tid >> 5;
  const int m0 = blockIdx.x * 64;
  if (tid < 64) {
    int row = m0 + tid;
    bool ok = row < M;
    sI[tid] = ok ? (idx0 ? idx0[row] : row) : 0;
    sI[64 + tid] = ok ? (idx1 ? idx1[row] : row) : 0;
  }
  __syncthreads();
  // gathered concat load, float4 granularity (96 float4 per row)
  for (int idx = tid; idx < 64 * 96; idx += 256) {
    int r = idx / 96, c4 = idx - r * 96;
    int row = m0 + r;
    float4 v = make_float4(0.f, 0.f, 0.f, 0.f);
    if (row < M) {
      if (c4 < 32)       v = ((const float4*)(X0 + (size_t)sI[r] * 128))[c4];
      else if (c4 < 64)  v = ((const float4*)(X1 + (size_t)sI[64 + r] * 128))[c4 - 32];
      else               v = ((const float4*)(X2 + (size_t)row * 128))[c4 - 64];
    }
    uint2 p; p.x = pk2(v.x, v.y); p.y = pk2(v.z, v.w);
    *(uint2*)(sX + r * 392 + c4 * 4) = p;
  }
  const int wM = wid >> 2, wN = wid & 3, hi8 = (lane >> 4) * 8;
  // ---- GEMM 1: [64,384] x [384,512], two 256-wide N chunks ----
  for (int nc = 0; nc < 2; ++nc) {
    v8f acc[2][4] = {{v8f_zero(), v8f_zero(), v8f_zero(), v8f_zero()},
                     {v8f_zero(), v8f_zero(), v8f_zero(), v8f_zero()}};
    for (int k0 = 0; k0 < 384; k0 += 32) {
      __syncthreads();
      for (int idx = tid; idx < 1024; idx += 256) {
        int nl = idx >> 2, q = idx & 3;
        const unsigned short* src = W1b + (size_t)(nc * 256 + nl) * 384 + k0;
        ((uint4*)(sW + nl * 40))[q] = ((const uint4*)src)[q];
        if (k0 + 32 < 384)
          __builtin_prefetch((const void*)((const uint4*)(src + 32) + q), 0, 0);
      }
      __syncthreads();
      v16bf a0 = load_frag(sX + (wM * 32) * 392 + k0, 392, lane);
      v16bf a1 = load_frag(sX + (wM * 32 + 16) * 392 + k0, 392, lane);
#pragma unroll
      for (int nt = 0; nt < 4; ++nt) {
        v16bf b = load_frag(sW + (wN * 64 + nt * 16) * 40, 40, lane);
        acc[0][nt] = wmma_bf16(a0, b, acc[0][nt]);
        acc[1][nt] = wmma_bf16(a1, b, acc[1][nt]);
      }
    }
#pragma unroll
    for (int mt = 0; mt < 2; ++mt)
#pragma unroll
      for (int nt = 0; nt < 4; ++nt)
#pragma unroll
        for (int i = 0; i < 8; ++i) {
          int r = wM * 32 + mt * 16 + i + hi8;
          int c = nc * 256 + wN * 64 + nt * 16 + (lane & 15);
          sF[r * 516 + c] = acc[mt][nt][i] + b1[c];
        }
  }
  __syncthreads();
  // ---- LN1 + SiLU -> bf16 hidden ----
  {
    int r = tid >> 2, seg = tid & 3;
    float s = 0.f, ss = 0.f;
    for (int c = seg * 128; c < seg * 128 + 128; ++c) {
      float v = sF[r * 516 + c];
      s += v; ss += v * v;
    }
    red[tid] = s; red[256 + tid] = ss;
  }
  __syncthreads();
  if (tid < 64) {
    float s = 0.f, ss = 0.f;
    for (int j = 0; j < 4; ++j) { s += red[tid * 4 + j]; ss += red[256 + tid * 4 + j]; }
    float mean = s * (1.f / 512.f), var = ss * (1.f / 512.f) - mean * mean;
    rstat[tid] = mean;
    rstat[64 + tid] = rsqrtf(var + 1e-5f);
  }
  __syncthreads();
  unsigned short* sHb = sX;  // [64][520]
  for (int idx = tid; idx < 64 * 256; idx += 256) {
    int r = idx >> 8, c = (idx & 255) * 2;
    float v0 = (sF[r * 516 + c] - rstat[r]) * rstat[64 + r] * g1[c] + be1[c];
    float v1 = (sF[r * 516 + c + 1] - rstat[r]) * rstat[64 + r] * g1[c + 1] + be1[c + 1];
    v0 = v0 / (1.f + __expf(-v0));
    v1 = v1 / (1.f + __expf(-v1));
    *(unsigned*)(sHb + r * 520 + c) = pk2(v0, v1);
  }
  // ---- GEMM 2: [64,512] x [512,128] ----
  v8f acc2[2][2] = {{v8f_zero(), v8f_zero()}, {v8f_zero(), v8f_zero()}};
  for (int k0 = 0; k0 < 512; k0 += 32) {
    __syncthreads();
    for (int idx = tid; idx < 512; idx += 256) {
      int nl = idx >> 2, q = idx & 3;
      const unsigned short* src = W2b + (size_t)nl * 512 + k0;
      ((uint4*)(sW + nl * 40))[q] = ((const uint4*)src)[q];
      if (k0 + 32 < 512)
        __builtin_prefetch((const void*)((const uint4*)(src + 32) + q), 0, 0);
    }
    __syncthreads();
    v16bf a0 = load_frag(sHb + (wM * 32) * 520 + k0, 520, lane);
    v16bf a1 = load_frag(sHb + (wM * 32 + 16) * 520 + k0, 520, lane);
    v16bf b0 = load_frag(sW + (wN * 32) * 40, 40, lane);
    v16bf b1 = load_frag(sW + (wN * 32 + 16) * 40, 40, lane);
    acc2[0][0] = wmma_bf16(a0, b0, acc2[0][0]);
    acc2[0][1] = wmma_bf16(a0, b1, acc2[0][1]);
    acc2[1][0] = wmma_bf16(a1, b0, acc2[1][0]);
    acc2[1][1] = wmma_bf16(a1, b1, acc2[1][1]);
  }
  __syncthreads();
#pragma unroll
  for (int mt = 0; mt < 2; ++mt)
#pragma unroll
    for (int nt = 0; nt < 2; ++nt)
#pragma unroll
      for (int i = 0; i < 8; ++i) {
        int r = wM * 32 + mt * 16 + i + hi8;
        int c = wN * 32 + nt * 16 + (lane & 15);
        sF[r * 136 + c] = acc2[mt][nt][i] + b2[c];
      }
  __syncthreads();
  // ---- LN2 + act + residual ----
  {
    int r = tid >> 2, seg = tid & 3;
    float s = 0.f, ss = 0.f;
    for (int c = seg * 32; c < seg * 32 + 32; ++c) {
      float v = sF[r * 136 + c];
      s += v; ss += v * v;
    }
    red[tid] = s; red[256 + tid] = ss;
  }
  __syncthreads();
  if (tid < 64) {
    float s = 0.f, ss = 0.f;
    for (int j = 0; j < 4; ++j) { s += red[tid * 4 + j]; ss += red[256 + tid * 4 + j]; }
    float mean = s * (1.f / 128.f), var = ss * (1.f / 128.f) - mean * mean;
    rstat[tid] = mean;
    rstat[64 + tid] = rsqrtf(var + 1e-5f);
  }
  __syncthreads();
  for (int idx = tid; idx < 64 * 128; idx += 256) {
    int r = idx >> 7, c = idx & 127;
    int row = m0 + r;
    if (row < M) {
      float v = (sF[r * 136 + c] - rstat[r]) * rstat[64 + r] * g2[c] + be2[c];
      float a = (act == 0) ? v / (1.f + __expf(-v)) : tanhf(v);
      Out[(size_t)row * 128 + c] = a + resid[(size_t)row * 128 + c];
    }
  }
}

// ---------------------------------------------------------------------------
// attn_keys256: MHA with exactly 256 keys resident in LDS.
// ---------------------------------------------------------------------------
__global__ void gnn_attn_keys256(const float* __restrict__ Q, const float* __restrict__ K,
                                 const float* __restrict__ V, float* __restrict__ O,
                                 int Lq) {
  extern __shared__ char smem[];
  unsigned short* sQ = (unsigned short*)smem;               // [64][40]
  unsigned short* sKb = (unsigned short*)(smem + 5120);     // [256][40]
  unsigned short* sVt = (unsigned short*)(smem + 25600);    // [16][264] (d-major)
  float* sS = (float*)(smem + 34048);                       // [64][260]
  unsigned short* sP = (unsigned short*)(smem + 100608);    // [64][264]
  float* red = (float*)(smem + 134400);                     // 512
  float* rstat = (float*)(smem + 136448);                   // 128
  const int tid = threadIdx.x, lane = tid & 31, wid = tid >> 5;
  const int h = blockIdx.y;
  const int q0 = blockIdx.x * 64;
  for (int idx = tid; idx < 64 * 16; idx += 256) {
    int r = idx >> 4, c = (idx & 15) * 2;
    int row = q0 + r;
    unsigned p = 0;
    if (c < 16 && row < Lq) {
      float2 v = *(const float2*)(Q + (size_t)row * 128 + h * 16 + c);
      p = pk2(v.x, v.y);
    }
    *(unsigned*)(sQ + r * 40 + c) = p;
  }
  for (int idx = tid; idx < 256 * 16; idx += 256) {
    int r = idx >> 4, c = (idx & 15) * 2;
    unsigned p = 0;
    if (c < 16) {
      float2 v = *(const float2*)(K + (size_t)r * 128 + h * 16 + c);
      p = pk2(v.x, v.y);
    }
    *(unsigned*)(sKb + r * 40 + c) = p;
  }
  for (int idx = tid; idx < 1024; idx += 256) {
    int k = idx >> 2, d4 = (idx & 3) * 4;
    float4 v = ((const float4*)(V + (size_t)k * 128 + h * 16))[idx & 3];
    sVt[(d4 + 0) * 264 + k] = f2bf(v.x);
    sVt[(d4 + 1) * 264 + k] = f2bf(v.y);
    sVt[(d4 + 2) * 264 + k] = f2bf(v.z);
    sVt[(d4 + 3) * 264 + k] = f2bf(v.w);
  }
  __syncthreads();
  {
    int wM = wid >> 2, wN = wid & 3;
    int hi8 = (lane >> 4) * 8;
    v16bf a0 = load_frag(sQ + (wM * 32) * 40, 40, lane);
    v16bf a1 = load_frag(sQ + (wM * 32 + 16) * 40, 40, lane);
#pragma unroll
    for (int nt = 0; nt < 4; ++nt) {
      v16bf b = load_frag(sKb + (wN * 64 + nt * 16) * 40, 40, lane);
      v8f s0 = wmma_bf16(a0, b, v8f_zero());
      v8f s1 = wmma_bf16(a1, b, v8f_zero());
#pragma unroll
      for (int i = 0; i < 8; ++i) {
        int c = wN * 64 + nt * 16 + (lane & 15);
        sS[(wM * 32 + i + hi8) * 260 + c] = s0[i];
        sS[(wM * 32 + 16 + i + hi8) * 260 + c] = s1[i];
      }
    }
  }
  __syncthreads();
  {
    int r = tid >> 2, seg = tid & 3;
    float mx = -3.0e38f;
    for (int c = seg * 64; c < seg * 64 + 64; ++c) mx = fmaxf(mx, sS[r * 260 + c]);
    red[tid] = mx;
  }
  __syncthreads();
  if (tid < 64) {
    float mx = -3.0e38f;
    for (int j = 0; j < 4; ++j) mx = fmaxf(mx, red[tid * 4 + j]);
    rstat[tid] = mx;
  }
  __syncthreads();
  {
    int r = tid >> 2, seg = tid & 3;
    float s = 0.f;
    for (int c = seg * 64; c < seg * 64 + 64; ++c) s += __expf(sS[r * 260 + c] - rstat[r]);
    red[tid] = s;
  }
  __syncthreads();
  if (tid < 64) {
    float s = 0.f;
    for (int j = 0; j < 4; ++j) s += red[tid * 4 + j];
    rstat[64 + tid] = 1.0f / s;
  }
  __syncthreads();
  for (int idx = tid; idx < 64 * 128; idx += 256) {
    int r = idx >> 7, c = (idx & 127) * 2;
    float e0 = __expf(sS[r * 260 + c] - rstat[r]) * rstat[64 + r];
    float e1 = __expf(sS[r * 260 + c + 1] - rstat[r]) * rstat[64 + r];
    *(unsigned*)(sP + r * 264 + c) = pk2(e0, e1);
  }
  __syncthreads();
  if (wid < 4) {
    v8f acc = v8f_zero();
#pragma unroll
    for (int kc = 0; kc < 8; ++kc) {
      v16bf a = load_frag(sP + (wid * 16) * 264 + kc * 32, 264, lane);
      v16bf b = load_frag(sVt + kc * 32, 264, lane);
      acc = wmma_bf16(a, b, acc);
    }
    int hi8 = (lane >> 4) * 8;
#pragma unroll
    for (int i = 0; i < 8; ++i) {
      int row = q0 + wid * 16 + i + hi8;
      if (row < Lq) O[(size_t)row * 128 + h * 16 + (lane & 15)] = acc[i];
    }
  }
}

// ---------------------------------------------------------------------------
// attn_stream: flash-style MHA, 32 queries/block, Lk streamed keys; 8 waves own
// disjoint 64-key chunks w/ private online-softmax state; LSE merge in LDS.
// ---------------------------------------------------------------------------
__global__ void gnn_attn_stream(const float* __restrict__ Q, const float* __restrict__ K,
                                const float* __restrict__ V, float* __restrict__ O,
                                int Lk) {
  extern __shared__ char smem[];
  const int tid = threadIdx.x, lane = tid & 31, wid = tid >> 5;
  unsigned short* sQ = (unsigned short*)smem;                                 // [32][40]
  unsigned short* sKb = (unsigned short*)(smem + 2560 + wid * 12032);         // [64][40]
  unsigned short* sVt = (unsigned short*)(smem + 2560 + wid * 12032 + 5120);  // [16][72]
  unsigned short* sP = (unsigned short*)(smem + 2560 + wid * 12032 + 7424);   // [32][72]
  float* cm = (float*)(smem + 98816);   // [8][32]
  float* cl = cm + 256;                 // [8][32]
  float* cO = cl + 256;                 // [8][32][16]
  float* rowM = cO + 4096;              // [32]
  float* rowL = rowM + 32;              // [32]
  const int h = blockIdx.y;
  const int q0 = blockIdx.x * 32;
  for (int idx = tid; idx < 32 * 16; idx += 256) {
    int r = idx >> 4, c = (idx & 15) * 2;
    unsigned p = 0;
    if (c < 16) {
      float2 v = *(const float2*)(Q + (size_t)(q0 + r) * 128 + h * 16 + c);
      p = pk2(v.x, v.y);
    }
    *(unsigned*)(sQ + r * 40 + c) = p;
  }
  // zero the k=16..31 pad of this wave's K staging
  for (int idx = lane; idx < 64 * 8; idx += 32) {
    int r = idx >> 3, c2 = idx & 7;
    *(unsigned*)(sKb + r * 40 + 16 + c2 * 2) = 0;
  }
  __syncthreads();
  float m[2][8], l[2][8];
  v8f Oa[2] = {v8f_zero(), v8f_zero()};
#pragma unroll
  for (int mt = 0; mt < 2; ++mt)
#pragma unroll
    for (int i = 0; i < 8; ++i) { m[mt][i] = -3.0e38f; l[mt][i] = 0.f; }

  const int hi8 = (lane >> 4) * 8;
  for (int k0 = wid * 64; k0 < Lk; k0 += 512) {
    for (int idx = lane; idx < 512; idx += 32) {
      int kk = idx >> 3, c = (idx & 7) * 2;
      int kg = k0 + kk;
      unsigned p = 0;
      if (kg < Lk) {
        float2 v = *(const float2*)(K + (size_t)kg * 128 + h * 16 + c);
        p = pk2(v.x, v.y);
      }
      *(unsigned*)(sKb + kk * 40 + c) = p;
    }
    for (int idx = lane; idx < 256; idx += 32) {
      int kk = idx >> 2, d4 = (idx & 3) * 4;
      int kg = k0 + kk;
      float4 v = make_float4(0.f, 0.f, 0.f, 0.f);
      if (kg < Lk) v = ((const float4*)(V + (size_t)kg * 128 + h * 16))[idx & 3];
      sVt[(d4 + 0) * 72 + kk] = f2bf(v.x);
      sVt[(d4 + 1) * 72 + kk] = f2bf(v.y);
      sVt[(d4 + 2) * 72 + kk] = f2bf(v.z);
      sVt[(d4 + 3) * 72 + kk] = f2bf(v.w);
    }
    v8f S[2][4];
    v16bf a0 = load_frag(sQ, 40, lane);
    v16bf a1 = load_frag(sQ + 16 * 40, 40, lane);
#pragma unroll
    for (int nt = 0; nt < 4; ++nt) {
      v16bf b = load_frag(sKb + nt * 16 * 40, 40, lane);
      S[0][nt] = wmma_bf16(a0, b, v8f_zero());
      S[1][nt] = wmma_bf16(a1, b, v8f_zero());
    }
#pragma unroll
    for (int mt = 0; mt < 2; ++mt) {
      float cmax[8];
#pragma unroll
      for (int i = 0; i < 8; ++i) cmax[i] = -3.0e38f;
#pragma unroll
      for (int nt = 0; nt < 4; ++nt) {
        bool ok = (k0 + nt * 16 + (lane & 15)) < Lk;
#pragma unroll
        for (int i = 0; i < 8; ++i) {
          float s = ok ? S[mt][nt][i] : -3.0e38f;
          S[mt][nt][i] = s;
          cmax[i] = fmaxf(cmax[i], s);
        }
      }
#pragma unroll
      for (int i = 0; i < 8; ++i) {
        float v = cmax[i];
        v = fmaxf(v, __shfl_xor(v, 1, 32));
        v = fmaxf(v, __shfl_xor(v, 2, 32));
        v = fmaxf(v, __shfl_xor(v, 4, 32));
        v = fmaxf(v, __shfl_xor(v, 8, 32));
        cmax[i] = v;
      }
#pragma unroll
      for (int i = 0; i < 8; ++i) {
        float mn = fmaxf(m[mt][i], cmax[i]);
        float sc = __expf(m[mt][i] - mn);
        float psum = 0.f;
#pragma unroll
        for (int nt = 0; nt < 4; ++nt) {
          float p = __expf(S[mt][nt][i] - mn);
          S[mt][nt][i] = p;
          psum += p;
        }
        psum += __shfl_xor(psum, 1, 32);
        psum += __shfl_xor(psum, 2, 32);
        psum += __shfl_xor(psum, 4, 32);
        psum += __shfl_xor(psum, 8, 32);
        l[mt][i] = l[mt][i] * sc + psum;
        m[mt][i] = mn;
        Oa[mt][i] *= sc;
      }
#pragma unroll
      for (int nt = 0; nt < 4; ++nt)
#pragma unroll
        for (int i = 0; i < 8; ++i)
          sP[(mt * 16 + i + hi8) * 72 + nt * 16 + (lane & 15)] = f2bf(S[mt][nt][i]);
    }
#pragma unroll
    for (int kc = 0; kc < 2; ++kc) {
      v16bf bv = load_frag(sVt + kc * 32, 72, lane);
#pragma unroll
      for (int mt = 0; mt < 2; ++mt) {
        v16bf ap = load_frag(sP + mt * 16 * 72 + kc * 32, 72, lane);
        Oa[mt] = wmma_bf16(ap, bv, Oa[mt]);
      }
    }
  }
#pragma unroll
  for (int mt = 0; mt < 2; ++mt)
#pragma unroll
    for (int i = 0; i < 8; ++i) {
      int r = mt * 16 + i + hi8;
      if ((lane & 15) == 0) { cm[wid * 32 + r] = m[mt][i]; cl[wid * 32 + r] = l[mt][i]; }
      cO[(wid * 32 + r) * 16 + (lane & 15)] = Oa[mt][i];
    }
  __syncthreads();
  if (tid < 32) {
    float M = -3.0e38f;
    for (int w = 0; w < 8; ++w) M = fmaxf(M, cm[w * 32 + tid]);
    float L = 0.f;
    for (int w = 0; w < 8; ++w) L += __expf(cm[w * 32 + tid] - M) * cl[w * 32 + tid];
    rowM[tid] = M;
    rowL[tid] = L;
  }
  __syncthreads();
  for (int idx = tid; idx < 512; idx += 256) {
    int r = idx >> 4, dd = idx & 15;
    float o = 0.f;
    for (int w = 0; w < 8; ++w)
      o += __expf(cm[w * 32 + r] - rowM[r]) * cO[(w * 32 + r) * 16 + dd];
    O[(size_t)(q0 + r) * 128 + h * 16 + dd] = o / rowL[r];
  }
}

// ---------------------------------------------------------------------------
__global__ void gnn_zero(float* __restrict__ p, int n) {
  int i = blockIdx.x * 256 + threadIdx.x;
  if (i < n) p[i] = 0.f;
}

__global__ void gnn_scatter_add(const float* __restrict__ e, const int* __restrict__ dst,
                                float* __restrict__ msg) {
  size_t i = (size_t)blockIdx.x * 256 + threadIdx.x;
  if (i < (size_t)NE * 128) {
    int ed = (int)(i >> 7), c = (int)(i & 127);
    atomicAdd(&msg[(size_t)dst[ed] * 128 + c], e[i]);
  }
}

// ---------------------------------------------------------------------------
extern "C" void kernel_launch(void* const* d_in, const int* in_sizes, int n_in,
                              void* d_out, int out_size, void* d_ws, size_t ws_size,
                              hipStream_t stream) {
  (void)in_sizes; (void)n_in; (void)out_size; (void)ws_size;
  const float* nodes = (const float*)d_in[0];
  const float* edges = (const float*)d_in[1];
  const float* supr  = (const float*)d_in[2];
  const int*   graph = (const int*)d_in[3];
  const float* em[8]; for (int i = 0; i < 8; ++i) em[i] = (const float*)d_in[4 + i];
  const float* nm[8]; for (int i = 0; i < 8; ++i) nm[i] = (const float*)d_in[12 + i];
  const float* sp[8]; for (int i = 0; i < 8; ++i) sp[i] = (const float*)d_in[20 + i];
  const float* up[4]; for (int i = 0; i < 4; ++i) up[i] = (const float*)d_in[28 + i];
  const float* sa[4]; for (int i = 0; i < 4; ++i) sa[i] = (const float*)d_in[32 + i];
  const float* dn[4]; for (int i = 0; i < 4; ++i) dn[i] = (const float*)d_in[36 + i];

  float* out_nodes = (float*)d_out;
  float* out_edges = out_nodes + (size_t)NN * 128;
  float* out_super = out_edges + (size_t)NE * 128;

  float* w = (float*)d_ws;
  float* Kn = w;      w += (size_t)NN * 128;
  float* Vn = w;      w += (size_t)NN * 128;
  float* Qn = w;      w += (size_t)NN * 128;
  float* Od = w;      w += (size_t)NN * 128;
  float* supmsg = w;  w += (size_t)NN * 128;
  float* edgemsg = w; w += (size_t)NN * 128;
  float* Qs = w;      w += NS * 128;
  float* Oup = w;     w += NS * 128;
  float* nodemsg = w; w += NS * 128;
  float* Qss = w;     w += NS * 128;
  float* Kss = w;     w += NS * 128;
  float* Vss = w;     w += NS * 128;
  float* Osa = w;     w += NS * 128;
  float* attnmsg = w; w += NS * 128;
  float* Ksd = w;     w += NS * 128;
  float* Vsd = w;     w += NS * 128;

  // bf16 weight copies (converted once per launch)
  unsigned short* wb = (unsigned short*)w;
  unsigned short* emW1b = wb; wb += 512 * 384;
  unsigned short* emW2b = wb; wb += 128 * 512;
  unsigned short* nmW1b = wb; wb += 512 * 384;
  unsigned short* nmW2b = wb; wb += 128 * 512;
  unsigned short* spW1b = wb; wb += 512 * 384;
  unsigned short* spW2b = wb; wb += 128 * 512;
  unsigned short* upWinb = wb;  wb += 384 * 128;
  unsigned short* upWoutb = wb; wb += 128 * 128;
  unsigned short* saWinb = wb;  wb += 384 * 128;
  unsigned short* saWoutb = wb; wb += 128 * 128;
  unsigned short* dnWinb = wb;  wb += 384 * 128;
  unsigned short* dnWoutb = wb; wb += 128 * 128;

  dim3 blk(256);
  const float qsc = 0.25f;  // hd^-0.5, hd = 16
  const int GB128 = (NN + 127) / 128;   // proj128 blocks over nodes
  const int GB64 = (NN + 63) / 64;      // attn_keys256 / mlp blocks over nodes

  // ---- weight conversion prepass ----
  struct Cv { const float* s; unsigned short* d; int n; };
  const Cv cv[12] = {
      {em[0], emW1b, 512 * 384}, {em[4], emW2b, 128 * 512},
      {nm[0], nmW1b, 512 * 384}, {nm[4], nmW2b, 128 * 512},
      {sp[0], spW1b, 512 * 384}, {sp[4], spW2b, 128 * 512},
      {up[0], upWinb, 384 * 128}, {up[2], upWoutb, 128 * 128},
      {sa[0], saWinb, 384 * 128}, {sa[2], saWoutb, 128 * 128},
      {dn[0], dnWinb, 384 * 128}, {dn[2], dnWoutb, 128 * 128},
  };
  for (int i = 0; i < 12; ++i) {
    int n4 = cv[i].n / 4;
    gnn_cvt_bf16<<<dim3((n4 + 255) / 256), blk, 0, stream>>>(cv[i].s, cv[i].d, n4);
  }

  // ---- supernode update: up cross-attn + self-attn + fused MLP ----
  gnn_proj128<<<dim3(2), blk, 0, stream>>>(supr, upWinb, up[1], Qs, NS, qsc);
  gnn_proj128<<<dim3(GB128), blk, 0, stream>>>(nodes, upWinb + 128 * 128, up[1] + 128, Kn, NN, 1.f);
  gnn_proj128<<<dim3(GB128), blk, 0, stream>>>(nodes, upWinb + 2 * 128 * 128, up[1] + 256, Vn, NN, 1.f);
  gnn_attn_stream<<<dim3(8, 8), blk, ATTN_STREAM_SMEM, stream>>>(Qs, Kn, Vn, Oup, NN);
  gnn_proj128<<<dim3(2), blk, 0, stream>>>(Oup, upWoutb, up[3], nodemsg, NS, 1.f);

  gnn_proj128<<<dim3(2), blk, 0, stream>>>(supr, saWinb, sa[1], Qss, NS, qsc);
  gnn_proj128<<<dim3(2), blk, 0, stream>>>(supr, saWinb + 128 * 128, sa[1] + 128, Kss, NS, 1.f);
  gnn_proj128<<<dim3(2), blk, 0, stream>>>(supr, saWinb + 2 * 128 * 128, sa[1] + 256, Vss, NS, 1.f);
  gnn_attn_keys256<<<dim3(4, 8), blk, ATTN256_SMEM, stream>>>(Qss, Kss, Vss, Osa, NS);
  gnn_proj128<<<dim3(2), blk, 0, stream>>>(Osa, saWoutb, sa[3], attnmsg, NS, 1.f);

  gnn_mlp_fused<<<dim3(4), blk, MLP_SMEM, stream>>>(
      supr, attnmsg, nodemsg, nullptr, nullptr, spW1b, sp[1], sp[2], sp[3], spW2b, sp[5],
      sp[6], sp[7], supr, out_super, NS, /*silu*/ 0);

  // ---- edge update: gather + fused MLP (tanh) + residual ----
  gnn_mlp_fused<<<dim3(NE / 64), blk, MLP_SMEM, stream>>>(
      nodes, nodes, edges, graph, graph + NE, emW1b, em[1], em[2], em[3], emW2b, em[5],
      em[6], em[7], edges, out_edges, NE, /*tanh*/ 1);

  // ---- node update: segment_sum + down cross-attn + fused MLP ----
  gnn_zero<<<dim3((NN * 128 + 255) / 256), blk, 0, stream>>>(edgemsg, NN * 128);
  gnn_scatter_add<<<dim3((int)(((size_t)NE * 128 + 255) / 256)), blk, 0, stream>>>(
      out_edges, graph + NE, edgemsg);

  gnn_proj128<<<dim3(GB128), blk, 0, stream>>>(nodes, dnWinb, dn[1], Qn, NN, qsc);
  gnn_proj128<<<dim3(2), blk, 0, stream>>>(out_super, dnWinb + 128 * 128, dn[1] + 128, Ksd, NS, 1.f);
  gnn_proj128<<<dim3(2), blk, 0, stream>>>(out_super, dnWinb + 2 * 128 * 128, dn[1] + 256, Vsd, NS, 1.f);
  gnn_attn_keys256<<<dim3(GB64, 8), blk, ATTN256_SMEM, stream>>>(Qn, Ksd, Vsd, Od, NN);
  gnn_proj128<<<dim3(GB128), blk, 0, stream>>>(Od, dnWoutb, dn[3], supmsg, NN, 1.f);

  gnn_mlp_fused<<<dim3(GB64), blk, MLP_SMEM, stream>>>(
      nodes, edgemsg, supmsg, nullptr, nullptr, nmW1b, nm[1], nm[2], nm[3], nmW2b, nm[5],
      nm[6], nm[7], nodes, out_nodes, NN, /*silu*/ 0);
}
